// PhaserModel_7464653160906
// MI455X (gfx1250) — matrix operand order repaired and checked by reading, not compiled, not run on hardware
//
#include <hip/hip_runtime.h>
#include <hip/hip_bf16.h>
#include <math.h>

// ---------------------------------------------------------------------------
// Phaser: LFO -> MLP(1->32->32->1) -> allpass coefficient p[n] -> 6-stage
// time-varying allpass cascade (serial recurrence, reformulated as a chunked
// affine scan since the per-sample update is affine in the 6-vector state z).
// ---------------------------------------------------------------------------

#define ORDER     6
#define HIDDEN    32
#define SR        44100.0f
#define MIN_RATE  0.1f
#define MAX_RATE  5.0f
#define TWO_PI    6.283185307179586f
#define CHUNK     256            // samples per scan chunk (pass1/pass3 depth)

typedef __attribute__((ext_vector_type(16))) _Float16 v16h;
typedef __attribute__((ext_vector_type(8)))  float    v8f;

// Raw v_rcp_f32 (single TRANS op; avoids the IEEE div_scale/div_fixup chain
// that __fdividef lowers to without -ffast-math).
__device__ __forceinline__ float fast_rcp(float x) {
    return __builtin_amdgcn_rcpf(x);
}

// tanh: gfx1250 has V_TANH_F32 in the TRANS pipe (co-executes with WMMA).
__device__ __forceinline__ float fast_tanh(float x) {
#if __has_builtin(__builtin_amdgcn_tanhf)
    return __builtin_amdgcn_tanhf(x);
#else
    // branch-free fallback: tanh(x) = 1 - 2/(e^{2x}+1)
    const float e = __expf(2.0f * x);
    return 1.0f - 2.0f * fast_rcp(e + 1.0f);
#endif
}
// Branch-free tan via native sin/cos + rcp (all TRANS ops).
__device__ __forceinline__ float fast_tan(float x) {
    return __sinf(x) * fast_rcp(__cosf(x));
}

// ===========================================================================
// Kernel 1: modulation path. One wave handles 16-sample row tiles.
//   A (16x32 f16): lane&15 = row M; lanes 0-15 hold K{0..7,16..23},
//                  lanes 16-31 hold K{8..15,24..31}  (ISA 7.12.2 layout)
//   B (32x16 f16): lane&15 = col N; same K split as A
//   C (16x16 f32): VGPR r: lanes 0-15 -> M=r, lanes 16-31 -> M=8+r; N=lane&15
// Two WMMAs per tile (cols 0-15 and 16-31 of W2). TRANS-heavy tanh work
// co-executes with the XDL WMMA pipe.
// ===========================================================================
__global__ __launch_bounds__(256) void phaser_mlp_wmma(
    const float* __restrict__ W1, const float* __restrict__ b1,
    const float* __restrict__ W2, const float* __restrict__ b2,
    const float* __restrict__ W3, const float* __restrict__ b3,
    const float* __restrict__ rate01, const float* __restrict__ stereo01,
    const float* __restrict__ amp_p, const float* __restrict__ bias_p,
    const float* __restrict__ depth_p, const float* __restrict__ phase_p,
    float* __restrict__ p_out, int nsamp, int ntiles)
{
    const int lane   = threadIdx.x & 31;
    const int wave   = blockIdx.x * (blockDim.x >> 5) + (threadIdx.x >> 5);
    const int nwaves = gridDim.x * (blockDim.x >> 5);

    const float rate = rate01[0] * (MAX_RATE - MIN_RATE) + MIN_RATE;
    const float w0   = (TWO_PI / SR) * rate;
    const float ph0  = phase_p[0];
    const float soff = stereo01[0] * TWO_PI;
    const float amp  = amp_p[0];
    const float bias = bias_p[0];
    const float dpth = depth_p[0];
    const float b3v  = b3[0];

    // Per-lane K subset of layer-1 weights (matches A-matrix half layout).
    float w1k[16], b1k[16];
#pragma unroll
    for (int i = 0; i < 16; ++i) {
        const int k = i + 8 * ((i >> 3) + (lane >> 4));
        w1k[i] = W1[k];
        b1k[i] = b1[k];
    }

    // W2 columns 0-15 and 16-31 in B-matrix layout (loaded once per wave).
    const int ncol = lane & 15;
    const int kgrp = (lane >> 4) << 4;
    v16h Bm0, Bm1;
#pragma unroll
    for (int i = 0; i < 16; ++i) {
        const int k = i + 8 * ((i >> 3) + (lane >> 4));
        Bm0[i] = (_Float16)W2[k * HIDDEN + ncol];
        Bm1[i] = (_Float16)W2[k * HIDDEN + 16 + ncol];
    }
    const float b2_0 = b2[ncol],      b2_1 = b2[16 + ncol];
    const float w3_0 = W3[ncol],      w3_1 = W3[16 + ncol];

    for (int tile = wave; tile < ntiles; tile += nwaves) {
        const int row = tile * 16 + (lane & 15);        // global (ch,n) row
        const int ch  = (row >= nsamp) ? 1 : 0;         // uniform per tile
        const int n   = row - ch * nsamp;
        const float arg = w0 * (float)(n + 1) + ph0 + (ch ? soff : 0.0f);
        const float lfo = amp * __cosf(arg);

        // h1 = tanh(lfo*W1 + b1), packed as 16x32 f16 A-matrix.
        v16h a;
#pragma unroll
        for (int i = 0; i < 16; ++i)
            a[i] = (_Float16)fast_tanh(lfo * w1k[i] + b1k[i]);

        v8f c0 = {}; v8f c1 = {};
        c0 = __builtin_amdgcn_wmma_f32_16x16x32_f16(false, a, false, Bm0,
                                                    (short)0, c0, false, false);
        c1 = __builtin_amdgcn_wmma_f32_16x16x32_f16(false, a, false, Bm1,
                                                    (short)0, c1, false, false);

        // h2 = tanh(c + b2); m = h2.W3 + b3 reduced over the 16-lane N group.
        float pv[8];
#pragma unroll
        for (int r = 0; r < 8; ++r) {
            float part = fast_tanh(c0[r] + b2_0) * w3_0
                       + fast_tanh(c1[r] + b2_1) * w3_1;
            part += __shfl_xor(part, 1);
            part += __shfl_xor(part, 2);
            part += __shfl_xor(part, 4);
            part += __shfl_xor(part, 8);
            const float mval = part + b3v;
            const float d    = bias + dpth * 0.5f * (1.0f + mval);
            const float td   = fast_tan(d);
            pv[r] = fast_tanh((1.0f - td) * fast_rcp(1.0f + td));
        }
        // lanes 0-15 hold rows 0..7, lanes 16-31 hold rows 8..15.
        if ((lane & 15) == 0) {
            const int base = tile * 16 + ((lane >> 4) << 3);
#pragma unroll
            for (int r = 0; r < 8; ++r) p_out[base + r] = pv[r];
        }
    }
}

// One allpass-cascade sample step on state z[6] (transposed DF-II, shared p).
__device__ __forceinline__ float ap_step(float z[ORDER], float x, float p) {
    float h = x;
#pragma unroll
    for (int k = 0; k < ORDER; ++k) {
        const float y = p * h + z[k];
        z[k] = h - p * y;
        h = y;
    }
    return h;   // cascade output before dry/wet mix
}

// ===========================================================================
// Kernel 2 (pass 1): per chunk, compose the per-sample affine maps
//   z' = M(p)z + g(p)x  into  (A_c, b_c)  by running the step on the 6
//   columns of A (x=0) and on b (x = actual input). Column-major A store.
// float4 streaming of p and x (chunk starts are 256-sample aligned).
// ===========================================================================
__global__ __launch_bounds__(256) void phaser_chunk_affine(
    const float* __restrict__ x, const float* __restrict__ p_ws,
    float* __restrict__ A_ws, float* __restrict__ b_ws,
    int nsamp, int nchunk_per_ch)
{
    const int c = blockIdx.x * blockDim.x + threadIdx.x;
    const int nct = 2 * nchunk_per_ch;
    if (c >= nct) return;
    const int ch    = c / nchunk_per_ch;
    const int coff  = (c % nchunk_per_ch) * CHUNK;
    const int start = ch * nsamp + coff;
    const int len   = min(CHUNK, nsamp - coff);

    float Acol[ORDER][ORDER];   // Acol[j][i] = A[i][j]
    float bvec[ORDER];
#pragma unroll
    for (int j = 0; j < ORDER; ++j) {
        bvec[j] = 0.0f;
#pragma unroll
        for (int i = 0; i < ORDER; ++i) Acol[j][i] = (i == j) ? 1.0f : 0.0f;
    }

    const float4* p4 = (const float4*)(p_ws + start);
    const float4* x4 = (const float4*)(x + start);
    int t = 0;
    for (; t + 3 < len; t += 4) {
        __builtin_prefetch(&p_ws[start + t + 128], 0, 1);
        __builtin_prefetch(&x[start + t + 128], 0, 1);
        const float4 pq = p4[t >> 2];
        const float4 xq = x4[t >> 2];
        const float ps[4] = {pq.x, pq.y, pq.z, pq.w};
        const float xs[4] = {xq.x, xq.y, xq.z, xq.w};
#pragma unroll
        for (int u = 0; u < 4; ++u) {
#pragma unroll
            for (int j = 0; j < ORDER; ++j) (void)ap_step(Acol[j], 0.0f, ps[u]);
            (void)ap_step(bvec, xs[u], ps[u]);
        }
    }
    for (; t < len; ++t) {                 // tail (len not multiple of 4)
        const float p  = p_ws[start + t];
        const float xv = x[start + t];
#pragma unroll
        for (int j = 0; j < ORDER; ++j) (void)ap_step(Acol[j], 0.0f, p);
        (void)ap_step(bvec, xv, p);
    }

#pragma unroll
    for (int j = 0; j < ORDER; ++j)
#pragma unroll
        for (int i = 0; i < ORDER; ++i)
            A_ws[c * 36 + j * 6 + i] = Acol[j][i];
#pragma unroll
    for (int i = 0; i < ORDER; ++i) b_ws[c * 6 + i] = bvec[i];
}

// ===========================================================================
// Kernel 3 (pass 2): short serial combine over chunk boundaries.
// Lane 0 -> channel 0, lane 1 -> channel 1. Stores each chunk's start state.
// ===========================================================================
__global__ __launch_bounds__(32) void phaser_chunk_combine(
    const float* __restrict__ A_ws, const float* __restrict__ b_ws,
    const float* __restrict__ zi, float* __restrict__ s_ws,
    int nchunk_per_ch)
{
    const int ch = threadIdx.x;
    if (ch >= 2) return;
    float s[ORDER];
#pragma unroll
    for (int k = 0; k < ORDER; ++k) s[k] = zi[ch * ORDER + k];

    for (int c = 0; c < nchunk_per_ch; ++c) {
        const int cc = ch * nchunk_per_ch + c;
#pragma unroll
        for (int k = 0; k < ORDER; ++k) s_ws[cc * 6 + k] = s[k];
        float sn[ORDER];
#pragma unroll
        for (int i = 0; i < ORDER; ++i) sn[i] = b_ws[cc * 6 + i];
#pragma unroll
        for (int j = 0; j < ORDER; ++j) {
            const float sj = s[j];
#pragma unroll
            for (int i = 0; i < ORDER; ++i)
                sn[i] += A_ws[cc * 36 + j * 6 + i] * sj;
        }
#pragma unroll
        for (int k = 0; k < ORDER; ++k) s[k] = sn[k];
    }
}

// ===========================================================================
// Kernel 4 (pass 3): replay each chunk from its known start state, emit
// y = 0.5*(x + cascade(x)). float4 streaming + vector stores.
// ===========================================================================
__global__ __launch_bounds__(256) void phaser_chunk_apply(
    const float* __restrict__ x, const float* __restrict__ p_ws,
    const float* __restrict__ s_ws, float* __restrict__ y,
    int nsamp, int nchunk_per_ch)
{
    const int c = blockIdx.x * blockDim.x + threadIdx.x;
    const int nct = 2 * nchunk_per_ch;
    if (c >= nct) return;
    const int ch    = c / nchunk_per_ch;
    const int coff  = (c % nchunk_per_ch) * CHUNK;
    const int start = ch * nsamp + coff;
    const int len   = min(CHUNK, nsamp - coff);

    float z[ORDER];
#pragma unroll
    for (int k = 0; k < ORDER; ++k) z[k] = s_ws[c * 6 + k];

    const float4* p4 = (const float4*)(p_ws + start);
    const float4* x4 = (const float4*)(x + start);
    float4*       y4 = (float4*)(y + start);
    int t = 0;
    for (; t + 3 < len; t += 4) {
        __builtin_prefetch(&p_ws[start + t + 128], 0, 1);
        __builtin_prefetch(&x[start + t + 128], 0, 1);
        const float4 pq = p4[t >> 2];
        const float4 xq = x4[t >> 2];
        float4 out;
        out.x = 0.5f * (xq.x + ap_step(z, xq.x, pq.x));
        out.y = 0.5f * (xq.y + ap_step(z, xq.y, pq.y));
        out.z = 0.5f * (xq.z + ap_step(z, xq.z, pq.z));
        out.w = 0.5f * (xq.w + ap_step(z, xq.w, pq.w));
        y4[t >> 2] = out;
    }
    for (; t < len; ++t) {                 // tail
        const float xv = x[start + t];
        const float h  = ap_step(z, xv, p_ws[start + t]);
        y[start + t] = 0.5f * (xv + h);
    }
}

// ===========================================================================
extern "C" void kernel_launch(void* const* d_in, const int* in_sizes, int n_in,
                              void* d_out, int out_size, void* d_ws, size_t ws_size,
                              hipStream_t stream) {
    const float* x        = (const float*)d_in[0];
    const float* rate01   = (const float*)d_in[1];
    const float* stereo01 = (const float*)d_in[2];
    const float* W1       = (const float*)d_in[3];
    const float* b1       = (const float*)d_in[4];
    const float* W2       = (const float*)d_in[5];
    const float* b2       = (const float*)d_in[6];
    const float* W3       = (const float*)d_in[7];
    const float* b3       = (const float*)d_in[8];
    const float* amp      = (const float*)d_in[9];
    const float* bias     = (const float*)d_in[10];
    const float* depth    = (const float*)d_in[11];
    const float* zi       = (const float*)d_in[12];
    const float* phase    = (const float*)d_in[13];
    float* y = (float*)d_out;

    const int nsamp = in_sizes[0] / 2;              // 262144
    const int rows  = 2 * nsamp;
    const int ntiles = rows / 16;
    const int nchunk_per_ch = (nsamp + CHUNK - 1) / CHUNK;
    const int nct = 2 * nchunk_per_ch;

    // workspace: p[rows] | A[nct*36] | b[nct*6] | s[nct*6]
    float* p_ws = (float*)d_ws;
    float* A_ws = p_ws + rows;
    float* b_ws = A_ws + (size_t)nct * 36;
    float* s_ws = b_ws + (size_t)nct * 6;

    phaser_mlp_wmma<<<256, 256, 0, stream>>>(
        W1, b1, W2, b2, W3, b3, rate01, stereo01, amp, bias, depth, phase,
        p_ws, nsamp, ntiles);

    phaser_chunk_affine<<<(nct + 255) / 256, 256, 0, stream>>>(
        x, p_ws, A_ws, b_ws, nsamp, nchunk_per_ch);

    phaser_chunk_combine<<<1, 32, 0, stream>>>(
        A_ws, b_ws, zi, s_ws, nchunk_per_ch);

    phaser_chunk_apply<<<(nct + 255) / 256, 256, 0, stream>>>(
        x, p_ws, s_ws, y, nsamp, nchunk_per_ch);
}
